// UniST_48309792145605
// MI455X (gfx1250) — matrix operand order repaired and checked by reading, not compiled, not run on hardware
//
#include <hip/hip_runtime.h>
#include <math.h>

typedef __bf16 bf16_t;
typedef __attribute__((ext_vector_type(16))) __bf16 v16bf;
typedef __attribute__((ext_vector_type(8)))  __bf16 v8bf;
typedef __attribute__((ext_vector_type(8)))  float  v8f;

#define DEPTH 12
#define HEADS 8
#define CDIM  512
#define HDIM  1024
#define SEQ   2048
#define BATCH 4
#define MTOT  (BATCH * SEQ)   /* 8192 token rows */
#define HD    64              /* head dim */

#define MODE_HEADS 0
#define MODE_VT    1
#define MODE_RESID 2
#define MODE_GELU  3

__device__ __forceinline__ bf16_t f2bf(float f) {
  unsigned u = __builtin_bit_cast(unsigned, f);
  u += 0x7FFFu + ((u >> 16) & 1u);            // round-to-nearest-even
  unsigned short s = (unsigned short)(u >> 16);
  return __builtin_bit_cast(bf16_t, s);
}
__device__ __forceinline__ v8f vzero8() {
  v8f z = {0.f, 0.f, 0.f, 0.f, 0.f, 0.f, 0.f, 0.f};
  return z;
}

// gfx1250 async memory->LDS copy (ASYNCcnt-tracked; bypasses VGPRs).
// vdst = LDS byte offset (low 32 bits of generic shared pointer, ISA 10.2),
// vaddr = 64-bit global address. 16 bytes per lane.
__device__ __forceinline__ void async_copy_b128(unsigned lds_off,
                                                const void* gptr) {
  unsigned long long ga = (unsigned long long)gptr;
  asm volatile("global_load_async_to_lds_b128 %0, %1, off"
               :: "v"(lds_off), "v"(ga) : "memory");
}
__device__ __forceinline__ void wait_async0() {
  asm volatile("s_wait_asynccnt 0x0" ::: "memory");
}

// ---------------------------------------------------------------------------
// Weight conversion: fp32 W[in=K][out=N] (batched over layers) -> bf16 Wt[N][K]
// ---------------------------------------------------------------------------
__global__ void wconv_kernel(const float* __restrict__ in, bf16_t* __restrict__ out,
                             int K, int N, size_t out_layer_stride) {
  int l = blockIdx.y;
  size_t total = (size_t)K * N;
  size_t idx = (size_t)blockIdx.x * blockDim.x + threadIdx.x;  // idx = n*K + kk
  if (idx >= total) return;
  int n  = (int)(idx / K);
  int kk = (int)(idx % K);
  out[(size_t)l * out_layer_stride + idx] =
      f2bf(in[(size_t)l * total + (size_t)kk * N + n]);
}

// ---------------------------------------------------------------------------
// LayerNorm: fp32 x[row][512] -> bf16 out[row][512]
// ---------------------------------------------------------------------------
__global__ __launch_bounds__(256) void ln_kernel(const float* __restrict__ x,
                                                 const float* __restrict__ w,
                                                 const float* __restrict__ b,
                                                 bf16_t* __restrict__ out) {
  int row = blockIdx.x, tid = threadIdx.x;
  const float* xr = x + (size_t)row * CDIM;
  float v0 = xr[tid], v1 = xr[tid + 256];
  __shared__ float red[256];
  red[tid] = v0 + v1;
  __syncthreads();
  for (int off = 128; off > 0; off >>= 1) {
    if (tid < off) red[tid] += red[tid + off];
    __syncthreads();
  }
  float mean = red[0] * (1.0f / CDIM);
  __syncthreads();
  float d0 = v0 - mean, d1 = v1 - mean;
  red[tid] = d0 * d0 + d1 * d1;
  __syncthreads();
  for (int off = 128; off > 0; off >>= 1) {
    if (tid < off) red[tid] += red[tid + off];
    __syncthreads();
  }
  float rstd = rsqrtf(red[0] * (1.0f / CDIM) + 1e-5f);
  bf16_t* orow = out + (size_t)row * CDIM;
  orow[tid]       = f2bf(d0 * rstd * w[tid] + b[tid]);
  orow[tid + 256] = f2bf(d1 * rstd * w[tid + 256] + b[tid + 256]);
}

// ---------------------------------------------------------------------------
// WMMA GEMM: C[M=8192][Nout] = A[M][K] * Bt[Nout][K]^T + bias, fused epilogue.
// Block = 256 threads (8 waves, 4x2); block tile 128x128; wave tile 32x64
// (2x4 WMMA accumulators). Double-buffered LDS filled by gfx1250 async
// memory->LDS copies (no VGPR staging); one barrier per K-step; prefetch k+2.
// ---------------------------------------------------------------------------
__global__ __launch_bounds__(256) void gemm_kernel(
    const bf16_t* __restrict__ A, const bf16_t* __restrict__ Bt,
    const float* __restrict__ bias, const float* resid,
    void* out, int Kdim, int Nout, float premul, int mode) {
  __shared__ __align__(32) bf16_t ldsA[2][128 * 32];
  __shared__ __align__(32) bf16_t ldsB[2][128 * 32];
  int tid  = threadIdx.x;
  int lane = tid & 31, wave = tid >> 5;
  int half = lane >> 4, colL = lane & 15;
  int wm = wave & 3, wn = wave >> 2;                 // 4(M) x 2(N) wave grid
  size_t mbase = (size_t)blockIdx.x * 128;
  size_t nbase = (size_t)blockIdx.y * 128;

  v8f acc[2][4];
#pragma unroll
  for (int i = 0; i < 2; ++i)
#pragma unroll
    for (int j = 0; j < 4; ++j) acc[i][j] = vzero8();

  int arow = tid >> 1, acol = (tid & 1) * 16;        // both tiles 128x32
  const bf16_t* Aptr = &A[(mbase + arow) * (size_t)Kdim + acol];
  const bf16_t* Bptr = &Bt[(nbase + arow) * (size_t)Kdim + acol];
  unsigned ldsAoff[2], ldsBoff[2];
#pragma unroll
  for (int bfi = 0; bfi < 2; ++bfi) {
    ldsAoff[bfi] = (unsigned)(uintptr_t)&ldsA[bfi][arow * 32 + acol];
    ldsBoff[bfi] = (unsigned)(uintptr_t)&ldsB[bfi][arow * 32 + acol];
  }

  // preload tile 0 via async copies (2 x b128 per 32B chunk)
  async_copy_b128(ldsAoff[0],      Aptr);
  async_copy_b128(ldsAoff[0] + 16, Aptr + 8);
  async_copy_b128(ldsBoff[0],      Bptr);
  async_copy_b128(ldsBoff[0] + 16, Bptr + 8);
  wait_async0();
  __syncthreads();

  int nsteps = Kdim >> 5;
  for (int step = 0; step < nsteps; ++step) {
    int cur = step & 1, nxt = cur ^ 1;
    bool more = (step + 1 < nsteps);
    if (more) {                 // kick off next-tile async copies early
      const bf16_t* ga = Aptr + (step + 1) * 32;
      const bf16_t* gb = Bptr + (step + 1) * 32;
      async_copy_b128(ldsAoff[nxt],      ga);
      async_copy_b128(ldsAoff[nxt] + 16, ga + 8);
      async_copy_b128(ldsBoff[nxt],      gb);
      async_copy_b128(ldsBoff[nxt] + 16, gb + 8);
    }
    if (step + 2 < nsteps) {    // -> global_prefetch_b8
      __builtin_prefetch(Aptr + (step + 2) * 32, 0, 3);
      __builtin_prefetch(Bptr + (step + 2) * 32, 0, 3);
    }

    v16bf afr[2], bfr[4];
#pragma unroll
    for (int i = 0; i < 2; ++i) {
      int r  = wm * 32 + i * 16 + colL;
      int kb = half * 16;
      const v8bf lo = *(const v8bf*)&ldsA[cur][r * 32 + kb];
      const v8bf hi = *(const v8bf*)&ldsA[cur][r * 32 + kb + 8];
#pragma unroll
      for (int e = 0; e < 8; ++e) { afr[i][e] = lo[e]; afr[i][e + 8] = hi[e]; }
    }
#pragma unroll
    for (int j = 0; j < 4; ++j) {
      int c = wn * 64 + j * 16 + colL;
      bfr[j] = *(const v16bf*)&ldsB[cur][c * 32 + half * 16];
    }
#pragma unroll
    for (int i = 0; i < 2; ++i)
#pragma unroll
      for (int j = 0; j < 4; ++j)
        acc[i][j] = __builtin_amdgcn_wmma_f32_16x16x32_bf16(
            false, afr[i], false, bfr[j], (short)0, acc[i][j], false, false);

    if (more) wait_async0();    // next tile resident before barrier
    __syncthreads();
  }

#pragma unroll
  for (int i = 0; i < 2; ++i) {
#pragma unroll
    for (int j = 0; j < 4; ++j) {
#pragma unroll
      for (int r = 0; r < 8; ++r) {
        int row = (int)mbase + wm * 32 + i * 16 + r + 8 * half;
        int col = (int)nbase + wn * 64 + j * 16 + colL;
        float v = acc[i][j][r] + bias[col];
        if (mode == MODE_HEADS) {
          int b = row >> 11, n = row & 2047, h = col >> 6, d = col & 63;
          ((bf16_t*)out)[(((size_t)b * HEADS + h) * SEQ + n) * HD + d] =
              f2bf(v * premul);
        } else if (mode == MODE_VT) {
          int b = row >> 11, n = row & 2047, h = col >> 6, d = col & 63;
          ((bf16_t*)out)[(((size_t)b * HEADS + h) * HD + d) * SEQ + n] = f2bf(v);
        } else if (mode == MODE_RESID) {
          ((float*)out)[(size_t)row * Nout + col] =
              resid[(size_t)row * Nout + col] + v;
        } else {  // MODE_GELU, exact erf gelu
          float g = 0.5f * v * (1.0f + erff(v * 0.70710678118654752f));
          ((bf16_t*)out)[(size_t)row * Nout + col] = f2bf(g);
        }
      }
    }
  }
}

// ---------------------------------------------------------------------------
// Flash attention: per (b,h); each wave owns 16 query rows, loops keys in
// blocks of 64 (one online-softmax pass per 16 WMMAs). Q is pre-scaled by
// hd^-0.5 in the QKV projection. q,k: [bh][n][64] bf16; vt: [bh][64][n].
// ---------------------------------------------------------------------------
__global__ __launch_bounds__(128) void attn_kernel(
    const bf16_t* __restrict__ q, const bf16_t* __restrict__ k,
    const bf16_t* __restrict__ vt, bf16_t* __restrict__ o) {
  int bh   = blockIdx.x;
  int tid  = threadIdx.x;
  int lane = tid & 31, wave = tid >> 5;
  int half = lane >> 4, colL = lane & 15;
  int qrow0 = blockIdx.y * 64 + wave * 16;
  const bf16_t* Q  = q  + (size_t)bh * SEQ * HD;
  const bf16_t* K  = k  + (size_t)bh * SEQ * HD;
  const bf16_t* VT = vt + (size_t)bh * HD * SEQ;
  __shared__ __align__(32) bf16_t ldsP[4][16 * 64];

  // Q fragments for both d-steps (reused across the whole key loop)
  v16bf qfr[2];
#pragma unroll
  for (int ds = 0; ds < 2; ++ds) {
    int kbA = ds * 32 + half * 16;
    const v8bf lo = *(const v8bf*)&Q[(size_t)(qrow0 + colL) * HD + kbA];
    const v8bf hi = *(const v8bf*)&Q[(size_t)(qrow0 + colL) * HD + kbA + 8];
#pragma unroll
    for (int e = 0; e < 8; ++e) { qfr[ds][e] = lo[e]; qfr[ds][e + 8] = hi[e]; }
  }

  v8f oacc[4];
#pragma unroll
  for (int j = 0; j < 4; ++j) oacc[j] = vzero8();
  float mrow[8], lrow[8];
#pragma unroll
  for (int r = 0; r < 8; ++r) { mrow[r] = -3.0e38f; lrow[r] = 0.f; }

  for (int kb = 0; kb < SEQ; kb += 64) {
    // S = Q K^T for a 16x64 score tile (8 WMMAs)
    v8f s[4];
#pragma unroll
    for (int kt = 0; kt < 4; ++kt) s[kt] = vzero8();
#pragma unroll
    for (int kt = 0; kt < 4; ++kt) {
      int key = kb + kt * 16 + colL;
#pragma unroll
      for (int ds = 0; ds < 2; ++ds) {
        v16bf kf = *(const v16bf*)&K[(size_t)key * HD + ds * 32 + half * 16];
        s[kt] = __builtin_amdgcn_wmma_f32_16x16x32_bf16(
            false, qfr[ds], false, kf, (short)0, s[kt], false, false);
      }
    }
    // online softmax update (row = element r + 8*half, cols across 16 lanes)
#pragma unroll
    for (int r = 0; r < 8; ++r) {
      float s0 = s[0][r], s1 = s[1][r], s2 = s[2][r], s3 = s[3][r];
      float mx = fmaxf(fmaxf(s0, s1), fmaxf(s2, s3));
#pragma unroll
      for (int mk = 1; mk < 16; mk <<= 1) mx = fmaxf(mx, __shfl_xor(mx, mk, 32));
      float nm = fmaxf(mrow[r], mx);
      float p0 = __expf(s0 - nm), p1 = __expf(s1 - nm);
      float p2 = __expf(s2 - nm), p3 = __expf(s3 - nm);
      float ps = (p0 + p1) + (p2 + p3);
#pragma unroll
      for (int mk = 1; mk < 16; mk <<= 1) ps += __shfl_xor(ps, mk, 32);
      float corr = __expf(mrow[r] - nm);
      mrow[r] = nm;
      lrow[r] = lrow[r] * corr + ps;
#pragma unroll
      for (int j = 0; j < 4; ++j) oacc[j][r] *= corr;
      int prow = (r + 8 * half) * 64 + colL;
      ldsP[wave][prow]      = f2bf(p0);
      ldsP[wave][prow + 16] = f2bf(p1);
      ldsP[wave][prow + 32] = f2bf(p2);
      ldsP[wave][prow + 48] = f2bf(p3);
    }
    // reload P (16x64) in A-fragment layout (wave-local LDS; DS ops in order)
    v16bf pf[2];
#pragma unroll
    for (int ks = 0; ks < 2; ++ks) {
      int kbA = ks * 32 + half * 16;
      const v8bf lo = *(const v8bf*)&ldsP[wave][colL * 64 + kbA];
      const v8bf hi = *(const v8bf*)&ldsP[wave][colL * 64 + kbA + 8];
#pragma unroll
      for (int e = 0; e < 8; ++e) { pf[ks][e] = lo[e]; pf[ks][e + 8] = hi[e]; }
    }
    // O += P V  (8 WMMAs)
#pragma unroll
    for (int j = 0; j < 4; ++j) {
      int d = j * 16 + colL;
#pragma unroll
      for (int ks = 0; ks < 2; ++ks) {
        v16bf vf =
            *(const v16bf*)&VT[(size_t)d * SEQ + kb + ks * 32 + half * 16];
        oacc[j] = __builtin_amdgcn_wmma_f32_16x16x32_bf16(
            false, pf[ks], false, vf, (short)0, oacc[j], false, false);
      }
    }
  }

  int b = bh >> 3, h = bh & 7;
#pragma unroll
  for (int j = 0; j < 4; ++j) {
#pragma unroll
    for (int r = 0; r < 8; ++r) {
      int n = qrow0 + r + 8 * half;
      size_t t = (size_t)b * SEQ + n;
      int c = h * HD + j * 16 + colL;
      o[t * CDIM + c] = f2bf(oacc[j][r] / lrow[r]);
    }
  }
}

// ---------------------------------------------------------------------------
// Host orchestration
// ---------------------------------------------------------------------------
extern "C" void kernel_launch(void* const* d_in, const int* in_sizes, int n_in,
                              void* d_out, int out_size, void* d_ws, size_t ws_size,
                              hipStream_t stream) {
  (void)in_sizes; (void)n_in; (void)out_size; (void)ws_size;
  const float* x_in = (const float*)d_in[0];
  const float* Wq = (const float*)d_in[1];
  const float* bq = (const float*)d_in[2];
  const float* Wk = (const float*)d_in[3];
  const float* bk = (const float*)d_in[4];
  const float* Wv = (const float*)d_in[5];
  const float* bv = (const float*)d_in[6];
  const float* Wo = (const float*)d_in[7];
  const float* bo = (const float*)d_in[8];
  const float* ln1w = (const float*)d_in[9];
  const float* ln1b = (const float*)d_in[10];
  const float* ln2w = (const float*)d_in[11];
  const float* ln2b = (const float*)d_in[12];
  const float* W1 = (const float*)d_in[13];
  const float* b1 = (const float*)d_in[14];
  const float* W2 = (const float*)d_in[15];
  const float* b2 = (const float*)d_in[16];

  // workspace layout (bytes): total ~126 MB
  char* ws = (char*)d_ws;
  const size_t CC = (size_t)CDIM * CDIM;      // 262144
  const size_t CH = (size_t)CDIM * HDIM;      // 524288
  const size_t LSTRIDE = 4 * CC + 2 * CH;     // bf16 elems per layer = 2097152
  bf16_t* wt = (bf16_t*)ws;      ws += DEPTH * LSTRIDE * sizeof(bf16_t);
  float*  xbuf = (float*)ws;     ws += (size_t)MTOT * CDIM * sizeof(float);
  bf16_t* hbuf = (bf16_t*)ws;    ws += (size_t)MTOT * CDIM * sizeof(bf16_t);
  bf16_t* qbuf = (bf16_t*)ws;    ws += (size_t)MTOT * CDIM * sizeof(bf16_t);
  bf16_t* kbuf = (bf16_t*)ws;    ws += (size_t)MTOT * CDIM * sizeof(bf16_t);
  bf16_t* vtbuf = (bf16_t*)ws;   ws += (size_t)MTOT * CDIM * sizeof(bf16_t);
  bf16_t* obuf = (bf16_t*)ws;    ws += (size_t)MTOT * CDIM * sizeof(bf16_t);
  bf16_t* mbuf = (bf16_t*)ws;    ws += (size_t)MTOT * HDIM * sizeof(bf16_t);

  const size_t OFF_WQ = 0, OFF_WK = CC, OFF_WV = 2 * CC, OFF_WO = 3 * CC;
  const size_t OFF_W1 = 4 * CC, OFF_W2 = 4 * CC + CH;

  // convert+transpose all weights to bf16 (batched over 12 layers per type)
  dim3 gCC((unsigned)(CC / 256), DEPTH);
  dim3 gCH((unsigned)(CH / 256), DEPTH);
  wconv_kernel<<<gCC, 256, 0, stream>>>(Wq, wt + OFF_WQ, CDIM, CDIM, LSTRIDE);
  wconv_kernel<<<gCC, 256, 0, stream>>>(Wk, wt + OFF_WK, CDIM, CDIM, LSTRIDE);
  wconv_kernel<<<gCC, 256, 0, stream>>>(Wv, wt + OFF_WV, CDIM, CDIM, LSTRIDE);
  wconv_kernel<<<gCC, 256, 0, stream>>>(Wo, wt + OFF_WO, CDIM, CDIM, LSTRIDE);
  wconv_kernel<<<gCH, 256, 0, stream>>>(W1, wt + OFF_W1, CDIM, HDIM, LSTRIDE);
  wconv_kernel<<<gCH, 256, 0, stream>>>(W2, wt + OFF_W2, HDIM, CDIM, LSTRIDE);

  hipMemcpyAsync(xbuf, x_in, (size_t)MTOT * CDIM * sizeof(float),
                 hipMemcpyDeviceToDevice, stream);

  dim3 g512(MTOT / 128, CDIM / 128);   // 64 x 4
  dim3 g1024(MTOT / 128, HDIM / 128);  // 64 x 8
  dim3 gattn(BATCH * HEADS, SEQ / 64);
  const float qscale = 0.125f;         // hd^-0.5

  for (int l = 0; l < DEPTH; ++l) {
    const bf16_t* wl = wt + (size_t)l * LSTRIDE;
    ln_kernel<<<MTOT, 256, 0, stream>>>(xbuf, ln1w + (size_t)l * CDIM,
                                        ln1b + (size_t)l * CDIM, hbuf);
    gemm_kernel<<<g512, 256, 0, stream>>>(hbuf, wl + OFF_WQ, bq + (size_t)l * CDIM,
                                          nullptr, qbuf, CDIM, CDIM, qscale,
                                          MODE_HEADS);
    gemm_kernel<<<g512, 256, 0, stream>>>(hbuf, wl + OFF_WK, bk + (size_t)l * CDIM,
                                          nullptr, kbuf, CDIM, CDIM, 1.0f,
                                          MODE_HEADS);
    gemm_kernel<<<g512, 256, 0, stream>>>(hbuf, wl + OFF_WV, bv + (size_t)l * CDIM,
                                          nullptr, vtbuf, CDIM, CDIM, 1.0f,
                                          MODE_VT);
    attn_kernel<<<gattn, 128, 0, stream>>>(qbuf, kbuf, vtbuf, obuf);
    gemm_kernel<<<g512, 256, 0, stream>>>(obuf, wl + OFF_WO, bo + (size_t)l * CDIM,
                                          xbuf, xbuf, CDIM, CDIM, 1.0f,
                                          MODE_RESID);
    ln_kernel<<<MTOT, 256, 0, stream>>>(xbuf, ln2w + (size_t)l * CDIM,
                                        ln2b + (size_t)l * CDIM, hbuf);
    gemm_kernel<<<g1024, 256, 0, stream>>>(hbuf, wl + OFF_W1, b1 + (size_t)l * HDIM,
                                           nullptr, mbuf, CDIM, HDIM, 1.0f,
                                           MODE_GELU);
    float* outx = (l == DEPTH - 1) ? (float*)d_out : xbuf;
    gemm_kernel<<<g512, 256, 0, stream>>>(mbuf, wl + OFF_W2, b2 + (size_t)l * CDIM,
                                          xbuf, outx, HDIM, CDIM, 1.0f,
                                          MODE_RESID);
  }
}
